// RESEGNN_37512244363761
// MI455X (gfx1250) — compile-verified, matrix operand order checked by reading, not run on hardware
//
#include <hip/hip_runtime.h>

// ---------------- constants from the reference ----------------
#define N_ENTS 100000
#define N_REL2 512
#define HDIM   256
#define TSTEPS 3
#define NEDGE  500000
#define MROWS  400000
#define RRELU_SLOPE (11.0f / 48.0f)

typedef float  v8f   __attribute__((ext_vector_type(8)));
typedef __bf16 v16bf __attribute__((ext_vector_type(16)));

// ---------------- helpers ----------------
__device__ __forceinline__ unsigned short f2bf(float f) {
    unsigned u = __float_as_uint(f);
    u += 0x7FFFu + ((u >> 16) & 1u);          // round-to-nearest-even
    return (unsigned short)(u >> 16);
}
__device__ __forceinline__ float sigmoidf(float x) {
    return 1.0f / (1.0f + expf(-x));
}

// ================= WMMA bf16 NT GEMM (double-buffered) =================
// C[M,N] = A[M,K] * Bt[N,K]^T  (+bias[n]) (+=C if accumulate)
// A,Bt bf16 row-major (K innermost), C f32.  K % 32 == 0.
// Block tile 128x128, 8 waves (4M x 2N), wave tile 32x64 = 2x4 WMMA frags.
#define TBM 128
#define TBN 128
#define TK  32
#define LDT 40   // padded ushort stride per tile row (bank-conflict free, 16B aligned)

__global__ __launch_bounds__(256) void gemm_bf16_nt(
    const unsigned short* __restrict__ A,
    const unsigned short* __restrict__ Bt,
    float* __restrict__ C,
    const float* __restrict__ bias,
    int M, int N, int K, int accumulate)
{
    __shared__ unsigned short sA[2][TBM * LDT];   // 2 x 10240 B
    __shared__ unsigned short sB[2][TBN * LDT];   // 2 x 10240 B

    const int tid   = threadIdx.x;
    const int wave  = tid >> 5;
    const int lane  = tid & 31;
    const int waveM = wave >> 1;       // 0..3 -> M offset *32
    const int waveN = wave & 1;        // 0..1 -> N offset *64
    const int m16   = lane & 15;
    const int hi    = lane >> 4;

    const int rowBase = blockIdx.y * TBM;
    const int colBase = blockIdx.x * TBN;

    v8f acc[2][4] = {};

    // global staging mapping: thread loads 16 bf16 of A and 16 bf16 of B
    const int r  = tid >> 1;            // tile row 0..127
    const int c  = (tid & 1) * 16;      // tile col {0,16}
    const int grow = rowBase + r;
    const int gcol = colBase + r;
    const bool aok = grow < M;
    const bool bok = gcol < N;
    const unsigned short* aptr = A  + (size_t)grow * K + c;
    const unsigned short* bptr = Bt + (size_t)gcol * K + c;

    // prologue: stage k0 = 0 into buffer 0
    {
        uint4 va0 = make_uint4(0,0,0,0), va1 = va0, vb0 = va0, vb1 = va0;
        if (aok) { const uint4* p = (const uint4*)(aptr); va0 = p[0]; va1 = p[1]; }
        if (bok) { const uint4* p = (const uint4*)(bptr); vb0 = p[0]; vb1 = p[1]; }
        *(uint4*)&sA[0][r * LDT + c]     = va0;
        *(uint4*)&sA[0][r * LDT + c + 8] = va1;
        *(uint4*)&sB[0][r * LDT + c]     = vb0;
        *(uint4*)&sB[0][r * LDT + c + 8] = vb1;
    }
    __syncthreads();

    int buf = 0;
    for (int k0 = 0; k0 < K; k0 += TK) {
        const bool has_next = (k0 + TK) < K;

        // 1) issue NEXT tile's global loads (in flight during the WMMAs below)
        uint4 va0 = make_uint4(0,0,0,0), va1 = va0, vb0 = va0, vb1 = va0;
        if (has_next) {
            if (aok) {
                const uint4* p = (const uint4*)(aptr + k0 + TK);
                va0 = p[0]; va1 = p[1];
                if (k0 + 2 * TK < K)   // -> global_prefetch_b8
                    __builtin_prefetch(aptr + k0 + 2 * TK, 0, 1);
            }
            if (bok) {
                const uint4* p = (const uint4*)(bptr + k0 + TK);
                vb0 = p[0]; vb1 = p[1];
            }
        }

        // 2) fragment loads from current buffer (documented 16-bit A/B layout)
        union Frag { v16bf v; unsigned int u[8]; };
        Frag a[2], b[4];
        #pragma unroll
        for (int mt = 0; mt < 2; ++mt) {
            const unsigned short* fb = &sA[buf][(waveM * 32 + mt * 16 + m16) * LDT + hi * 8];
            #pragma unroll
            for (int v = 0; v < 8; ++v) {
                int kb = (v & 3) * 2 + (v >> 2) * 16;   // K pairs {0,2,4,6,16,18,20,22}+hi*8
                a[mt].u[v] = *(const unsigned int*)(fb + kb);
            }
        }
        #pragma unroll
        for (int nt = 0; nt < 4; ++nt) {
            const unsigned short* fb = &sB[buf][(waveN * 64 + nt * 16 + m16) * LDT + hi * 8];
            #pragma unroll
            for (int v = 0; v < 8; ++v) {
                int kb = (v & 3) * 2 + (v >> 2) * 16;
                b[nt].u[v] = *(const unsigned int*)(fb + kb);
            }
        }

        // 3) 8 WMMAs (global loads for next tile still in flight)
        #pragma unroll
        for (int mt = 0; mt < 2; ++mt)
            #pragma unroll
            for (int nt = 0; nt < 4; ++nt)
                acc[mt][nt] = __builtin_amdgcn_wmma_f32_16x16x32_bf16(
                    false, a[mt].v, false, b[nt].v, (short)0, acc[mt][nt], false, false);

        // 4) stage next tile into the alternate buffer
        if (has_next) {
            *(uint4*)&sA[buf ^ 1][r * LDT + c]     = va0;
            *(uint4*)&sA[buf ^ 1][r * LDT + c + 8] = va1;
            *(uint4*)&sB[buf ^ 1][r * LDT + c]     = vb0;
            *(uint4*)&sB[buf ^ 1][r * LDT + c + 8] = vb1;
        }
        __syncthreads();
        buf ^= 1;
    }

    // ---- epilogue: C layout VGPR r -> M = r + 8*hi, N = lane&15 ----
    #pragma unroll
    for (int mt = 0; mt < 2; ++mt) {
        #pragma unroll
        for (int nt = 0; nt < 4; ++nt) {
            int col = colBase + waveN * 64 + nt * 16 + m16;
            if (col >= N) continue;
            float bv = bias ? bias[col] : 0.0f;
            #pragma unroll
            for (int rr = 0; rr < 8; ++rr) {
                int row = rowBase + waveM * 32 + mt * 16 + rr + hi * 8;
                if (row < M) {
                    size_t idx = (size_t)row * N + col;
                    float val = acc[mt][nt][rr] + bv;
                    if (accumulate) val += C[idx];
                    C[idx] = val;
                }
            }
        }
    }
}

// ================= elementwise / reduction kernels =================

// wave-per-row: optional rrelu then L2 normalize; writes f32 + bf16 shadow
__global__ __launch_bounds__(256) void rrelu_norm_kernel(
    const float* __restrict__ in, float* __restrict__ out,
    unsigned short* __restrict__ out16, int rows, int do_rrelu)
{
    int wid = blockIdx.x * 8 + (threadIdx.x >> 5);
    if (wid >= rows) return;
    int lane = threadIdx.x & 31;
    const float* row = in + (size_t)wid * HDIM;
    float v[8]; float ss = 0.0f;
    #pragma unroll
    for (int i = 0; i < 8; ++i) {
        float x = row[lane + i * 32];
        if (do_rrelu) x = (x >= 0.0f) ? x : x * RRELU_SLOPE;
        v[i] = x; ss += x * x;
    }
    #pragma unroll
    for (int o = 16; o >= 1; o >>= 1) ss += __shfl_xor(ss, o, 32);
    float inv = 1.0f / fmaxf(sqrtf(ss), 1e-12f);
    size_t base = (size_t)wid * HDIM;
    #pragma unroll
    for (int i = 0; i < 8; ++i) {
        float x = v[i] * inv;
        out[base + lane + i * 32] = x;
        out16[base + lane + i * 32] = f2bf(x);
    }
}

// curH = l2norm(rrelu(agg/max(deg,1) + he_self))   (wave per row)
__global__ __launch_bounds__(256) void combine_msg_norm_kernel(
    const float* __restrict__ agg, const float* __restrict__ deg,
    const float* __restrict__ he_self, float* __restrict__ out,
    unsigned short* __restrict__ out16, int rows)
{
    int wid = blockIdx.x * 8 + (threadIdx.x >> 5);
    if (wid >= rows) return;
    int lane = threadIdx.x & 31;
    float dinv = 1.0f / fmaxf(deg[wid], 1.0f);
    size_t base = (size_t)wid * HDIM;
    float v[8]; float ss = 0.0f;
    #pragma unroll
    for (int i = 0; i < 8; ++i) {
        int c = lane + i * 32;
        float x = agg[base + c] * dinv + he_self[base + c];
        x = (x >= 0.0f) ? x : x * RRELU_SLOPE;
        v[i] = x; ss += x * x;
    }
    #pragma unroll
    for (int o = 16; o >= 1; o >>= 1) ss += __shfl_xor(ss, o, 32);
    float inv = 1.0f / fmaxf(sqrtf(ss), 1e-12f);
    #pragma unroll
    for (int i = 0; i < 8; ++i) {
        float x = v[i] * inv;
        out[base + lane + i * 32] = x;
        out16[base + lane + i * 32] = f2bf(x);
    }
}

// GRU gate math (torch GRUCell); optional row L2 norm (wave per row)
__global__ __launch_bounds__(256) void gru_elem_kernel(
    const float* __restrict__ gi, const float* __restrict__ gh,
    const float* __restrict__ hprev, float* __restrict__ out,
    unsigned short* __restrict__ out16, int rows, int do_norm)
{
    int wid = blockIdx.x * 8 + (threadIdx.x >> 5);
    if (wid >= rows) return;
    int lane = threadIdx.x & 31;
    const float* gir = gi + (size_t)wid * (3 * HDIM);
    const float* ghr = gh + (size_t)wid * (3 * HDIM);
    const float* hr  = hprev + (size_t)wid * HDIM;
    float h[8]; float ss = 0.0f;
    #pragma unroll
    for (int i = 0; i < 8; ++i) {
        int c = lane + i * 32;
        float rg = sigmoidf(gir[c] + ghr[c]);
        float z  = sigmoidf(gir[HDIM + c] + ghr[HDIM + c]);
        float n  = tanhf(gir[2 * HDIM + c] + rg * ghr[2 * HDIM + c]);
        float hv = (1.0f - z) * n + z * hr[c];
        h[i] = hv; ss += hv * hv;
    }
    if (do_norm) {
        #pragma unroll
        for (int o = 16; o >= 1; o >>= 1) ss += __shfl_xor(ss, o, 32);
        float inv = 1.0f / fmaxf(sqrtf(ss), 1e-12f);
        #pragma unroll
        for (int i = 0; i < 8; ++i) h[i] *= inv;
    }
    size_t base = (size_t)wid * HDIM;
    #pragma unroll
    for (int i = 0; i < 8; ++i) {
        out[base + lane + i * 32] = h[i];
        out16[base + lane + i * 32] = f2bf(h[i]);
    }
}

// per-relation segment mean using sorted r_seg (binary search, no atomics)
__global__ __launch_bounds__(256) void seg_mean_kernel(
    const float* __restrict__ he, const int* __restrict__ r_to_e,
    const int* __restrict__ r_seg, float* __restrict__ x_f32,
    unsigned short* __restrict__ x_b16)
{
    int b = blockIdx.x;   // relation id
    int lo = 0, hiB = MROWS;
    while (lo < hiB) { int mid = (lo + hiB) >> 1; if (r_seg[mid] < b) lo = mid + 1; else hiB = mid; }
    int lo2 = lo, hi2 = MROWS;
    while (lo2 < hi2) { int mid = (lo2 + hi2) >> 1; if (r_seg[mid] < b + 1) lo2 = mid + 1; else hi2 = mid; }
    int c = threadIdx.x;
    float s = 0.0f;
    for (int j = lo; j < lo2; ++j)
        s += he[(size_t)r_to_e[j] * HDIM + c];
    float cnt = (float)(lo2 - lo);
    float o = s / fmaxf(cnt, 1.0f);
    x_f32[b * HDIM + c] = o;
    x_b16[b * HDIM + c] = f2bf(o);
}

// edge scatter: agg[dst] += he_msg[src] + r_msg[rel]; deg[dst]++
__global__ __launch_bounds__(256) void edge_agg_kernel(
    const int* __restrict__ src, const int* __restrict__ dst,
    const int* __restrict__ rel, const float* __restrict__ he_msg,
    const float* __restrict__ r_msg, float* __restrict__ agg,
    float* __restrict__ deg)
{
    int e = blockIdx.x;
    int s = src[e], d = dst[e], rr = rel[e];
    int c = threadIdx.x;
    float v = he_msg[(size_t)s * HDIM + c] + r_msg[(size_t)rr * HDIM + c];
    atomicAdd(&agg[(size_t)d * HDIM + c], v);
    if (c == 0) atomicAdd(&deg[d], 1.0f);
}

// out = tw*a + (1-tw)*b where tw = sigmoid(gate); writes f32, bf16, and d_out copy
__global__ __launch_bounds__(256) void gate_mix_kernel(
    const float* __restrict__ gate, const float* __restrict__ a,
    const float* __restrict__ b, float* __restrict__ out,
    unsigned short* __restrict__ out16, float* __restrict__ out_copy,
    long long n)
{
    long long i = (long long)blockIdx.x * blockDim.x + threadIdx.x;
    long long stride = (long long)gridDim.x * blockDim.x;
    for (; i < n; i += stride) {
        float tw = sigmoidf(gate[i]);
        float v = tw * a[i] + (1.0f - tw) * b[i];
        out[i] = v;
        out16[i] = f2bf(v);
        out_copy[i] = v;
    }
}

__global__ __launch_bounds__(256) void zero_f32_kernel(float* __restrict__ p, long long n)
{
    long long i = (long long)blockIdx.x * blockDim.x + threadIdx.x;
    long long stride = (long long)gridDim.x * blockDim.x;
    for (; i < n; i += stride) p[i] = 0.0f;
}

__global__ __launch_bounds__(256) void copy_cast_kernel(
    const float* __restrict__ in, float* __restrict__ out,
    unsigned short* __restrict__ out16, long long n)
{
    long long i = (long long)blockIdx.x * blockDim.x + threadIdx.x;
    long long stride = (long long)gridDim.x * blockDim.x;
    for (; i < n; i += stride) { float v = in[i]; out[i] = v; out16[i] = f2bf(v); }
}

// Bt[n*K+k] = bf16(in[k*N+n])  (for x @ W with W [K,N])
__global__ __launch_bounds__(256) void cast_transpose_kernel(
    const float* __restrict__ in, unsigned short* __restrict__ out, int K, int N)
{
    int idx = blockIdx.x * 256 + threadIdx.x;
    if (idx >= K * N) return;
    int n = idx / K, k = idx % K;
    out[idx] = f2bf(in[(size_t)k * N + n]);
}

// direct cast (GRU weights already [N,K] for x @ W.T)
__global__ __launch_bounds__(256) void cast_direct_kernel(
    const float* __restrict__ in, unsigned short* __restrict__ out, long long n)
{
    long long i = (long long)blockIdx.x * blockDim.x + threadIdx.x;
    long long stride = (long long)gridDim.x * blockDim.x;
    for (; i < n; i += stride) out[i] = f2bf(in[i]);
}

// ================= host side =================
static inline void launch_gemm(const unsigned short* A, const unsigned short* Bt,
                               float* C, const float* bias, int M, int N, int K,
                               int acc, hipStream_t s)
{
    dim3 g((N + TBN - 1) / TBN, (M + TBM - 1) / TBM);
    gemm_bf16_nt<<<g, 256, 0, s>>>(A, Bt, C, bias, M, N, K, acc);
}

extern "C" void kernel_launch(void* const* d_in, const int* in_sizes, int n_in,
                              void* d_out, int out_size, void* d_ws, size_t ws_size,
                              hipStream_t stream)
{
    (void)in_sizes; (void)n_in; (void)out_size; (void)ws_size;
    const float* dynamic_emb = (const float*)d_in[0];
    const float* emb_rel     = (const float*)d_in[1];
    const float* w_msg       = (const float*)d_in[2];
    const float* w_self      = (const float*)d_in[3];
    const float* w_rel       = (const float*)d_in[4];
    const float* e_gate_w    = (const float*)d_in[5];
    const float* e_gate_b    = (const float*)d_in[6];
    const float* r_gate_w    = (const float*)d_in[7];
    const float* r_gate_b    = (const float*)d_in[8];
    const float* gru_r_wih   = (const float*)d_in[9];
    const float* gru_r_whh   = (const float*)d_in[10];
    const float* gru_r_bih   = (const float*)d_in[11];
    const float* gru_r_bhh   = (const float*)d_in[12];
    const float* gru_e_wih   = (const float*)d_in[13];
    const float* gru_e_whh   = (const float*)d_in[14];
    const float* gru_e_bih   = (const float*)d_in[15];
    const float* gru_e_bhh   = (const float*)d_in[16];
    const int*   src         = (const int*)d_in[17];
    const int*   dst         = (const int*)d_in[18];
    const int*   rel         = (const int*)d_in[19];
    const int*   r_to_e      = (const int*)d_in[20];
    const int*   r_seg       = (const int*)d_in[21];
    float* out = (float*)d_out;

    // ---- workspace layout ----
    char* base = (char*)d_ws;
    size_t off = 0;
    auto alloc = [&](size_t bytes) -> void* {
        void* p = base + off;
        off = (off + bytes + 255) & ~(size_t)255;
        return p;
    };
    const size_t NH   = (size_t)N_ENTS * HDIM;
    const size_t RH   = (size_t)N_REL2 * HDIM;
    float*          he_f32   = (float*)alloc(NH * 4);
    unsigned short* he_b16   = (unsigned short*)alloc(NH * 2);
    float*          curh_f32 = (float*)alloc(NH * 4);
    unsigned short* curh_b16 = (unsigned short*)alloc(NH * 2);
    float*          hnew_f32 = (float*)alloc(NH * 4);
    unsigned short* hnew_b16 = (unsigned short*)alloc(NH * 2);
    float*          he_msg   = (float*)alloc(NH * 4);   // also reused as gate buffer
    float*          he_self  = (float*)alloc(NH * 4);
    float*          agg      = (float*)alloc((NH + N_ENTS) * 4);
    float*          deg      = agg + NH;
    float*          gi       = (float*)alloc(NH * 3 * 4);
    float*          gh       = (float*)alloc(NH * 3 * 4);
    float*          hr_f32   = (float*)alloc(RH * 4);
    unsigned short* hr_b16   = (unsigned short*)alloc(RH * 2);
    float*          x_f32    = (float*)alloc(RH * 4);
    unsigned short* x_b16    = (unsigned short*)alloc(RH * 2);
    float*          r_gi     = (float*)alloc(RH * 3 * 4);
    float*          r_gh     = (float*)alloc(RH * 3 * 4);
    float*          curR_f32 = (float*)alloc(RH * 4);
    unsigned short* curR_b16 = (unsigned short*)alloc(RH * 2);
    float*          r2_tmp   = (float*)alloc(RH * 4);
    float*          curR2_f32= (float*)alloc(RH * 4);
    unsigned short* curR2_b16= (unsigned short*)alloc(RH * 2);
    float*          r_msg    = (float*)alloc(RH * 4);
    float*          rg       = (float*)alloc(RH * 4);
    unsigned short* wmsgT    = (unsigned short*)alloc((size_t)HDIM * HDIM * 2);
    unsigned short* wselfT   = (unsigned short*)alloc((size_t)HDIM * HDIM * 2);
    unsigned short* wrelT    = (unsigned short*)alloc((size_t)HDIM * HDIM * 2);
    unsigned short* egateT   = (unsigned short*)alloc((size_t)HDIM * HDIM * 2);
    unsigned short* rgateT   = (unsigned short*)alloc((size_t)HDIM * HDIM * 2);
    unsigned short* WihR     = (unsigned short*)alloc((size_t)3 * HDIM * HDIM * 2);
    unsigned short* WhhR     = (unsigned short*)alloc((size_t)3 * HDIM * HDIM * 2);
    unsigned short* WihE     = (unsigned short*)alloc((size_t)3 * HDIM * HDIM * 2);
    unsigned short* WhhE     = (unsigned short*)alloc((size_t)3 * HDIM * HDIM * 2);

    // ---- one-time prep (redone each call; deterministic) ----
    int tpb = 256;
    int wgrid_e = (N_ENTS + 7) / 8;
    int wgrid_r = (N_REL2 + 7) / 8;
    // h_e = l2norm(dynamic_emb)
    rrelu_norm_kernel<<<wgrid_e, tpb, 0, stream>>>(dynamic_emb, he_f32, he_b16, N_ENTS, 0);
    // h_r = emb_rel
    copy_cast_kernel<<<512, tpb, 0, stream>>>(emb_rel, hr_f32, hr_b16, (long long)RH);
    // weights -> bf16 NT layout
    int hh = HDIM * HDIM;
    cast_transpose_kernel<<<(hh + 255) / 256, tpb, 0, stream>>>(w_msg,    wmsgT,  HDIM, HDIM);
    cast_transpose_kernel<<<(hh + 255) / 256, tpb, 0, stream>>>(w_self,   wselfT, HDIM, HDIM);
    cast_transpose_kernel<<<(hh + 255) / 256, tpb, 0, stream>>>(w_rel,    wrelT,  HDIM, HDIM);
    cast_transpose_kernel<<<(hh + 255) / 256, tpb, 0, stream>>>(e_gate_w, egateT, HDIM, HDIM);
    cast_transpose_kernel<<<(hh + 255) / 256, tpb, 0, stream>>>(r_gate_w, rgateT, HDIM, HDIM);
    long long whn = (long long)3 * HDIM * HDIM;
    cast_direct_kernel<<<768, tpb, 0, stream>>>(gru_r_wih, WihR, whn);
    cast_direct_kernel<<<768, tpb, 0, stream>>>(gru_r_whh, WhhR, whn);
    cast_direct_kernel<<<768, tpb, 0, stream>>>(gru_e_wih, WihE, whn);
    cast_direct_kernel<<<768, tpb, 0, stream>>>(gru_e_whh, WhhE, whn);

    for (int t = 0; t < TSTEPS; ++t) {
        const int* src_t = src + (size_t)t * NEDGE;
        const int* dst_t = dst + (size_t)t * NEDGE;
        const int* rel_t = rel + (size_t)t * NEDGE;
        const int* rte_t = r_to_e + (size_t)t * MROWS;
        const int* seg_t = r_seg + (size_t)t * MROWS;

        // 1) per-relation segment mean of touched entities
        seg_mean_kernel<<<N_REL2, tpb, 0, stream>>>(he_f32, rte_t, seg_t, x_f32, x_b16);
        // 2) relation GRU
        launch_gemm(x_b16,  WihR, r_gi, gru_r_bih, N_REL2, 3 * HDIM, HDIM, 0, stream);
        launch_gemm(hr_b16, WhhR, r_gh, gru_r_bhh, N_REL2, 3 * HDIM, HDIM, 0, stream);
        gru_elem_kernel<<<wgrid_r, tpb, 0, stream>>>(r_gi, r_gh, hr_f32, curR_f32, curR_b16, N_REL2, 0);
        // 3) distributed message GEMMs (h_e+r)@W = h_e@W + r@W
        launch_gemm(he_b16,   wmsgT,  he_msg,  nullptr, N_ENTS, HDIM, HDIM, 0, stream);
        launch_gemm(curR_b16, wmsgT,  r_msg,   nullptr, N_REL2, HDIM, HDIM, 0, stream);
        launch_gemm(he_b16,   wselfT, he_self, nullptr, N_ENTS, HDIM, HDIM, 0, stream);
        // 4) edge scatter-add (L2-resident atomics)
        zero_f32_kernel<<<4096, tpb, 0, stream>>>(agg, (long long)(NH + N_ENTS));
        edge_agg_kernel<<<NEDGE, tpb, 0, stream>>>(src_t, dst_t, rel_t, he_msg, r_msg, agg, deg);
        // 5) current_h = l2norm(rrelu(agg/deg + h_e@w_self))
        combine_msg_norm_kernel<<<wgrid_e, tpb, 0, stream>>>(agg, deg, he_self, curh_f32, curh_b16, N_ENTS);
        // 6) current_r2 = l2norm(rrelu(current_r @ w_rel))
        launch_gemm(curR_b16, wrelT, r2_tmp, nullptr, N_REL2, HDIM, HDIM, 0, stream);
        rrelu_norm_kernel<<<wgrid_r, tpb, 0, stream>>>(r2_tmp, curR2_f32, curR2_b16, N_REL2, 1);
        // 7) entity GRU + l2norm
        launch_gemm(curh_b16, WihE, gi, gru_e_bih, N_ENTS, 3 * HDIM, HDIM, 0, stream);
        launch_gemm(he_b16,   WhhE, gh, gru_e_bhh, N_ENTS, 3 * HDIM, HDIM, 0, stream);
        gru_elem_kernel<<<wgrid_e, tpb, 0, stream>>>(gi, gh, he_f32, hnew_f32, hnew_b16, N_ENTS, 1);
        // 8) entity gate mix -> new h_e, history[t]
        launch_gemm(hnew_b16, egateT, he_msg /*gate buf*/, e_gate_b, N_ENTS, HDIM, HDIM, 0, stream);
        gate_mix_kernel<<<8192, tpb, 0, stream>>>(he_msg, curh_f32, hnew_f32,
                                                  he_f32, he_b16,
                                                  out + (size_t)t * NH, (long long)NH);
        // 9) relation gate mix -> new h_r (tail of output; last t wins)
        launch_gemm(curR2_b16, rgateT, rg, r_gate_b, N_REL2, HDIM, HDIM, 0, stream);
        gate_mix_kernel<<<512, tpb, 0, stream>>>(rg, curR2_f32, hr_f32,
                                                 hr_f32, hr_b16,
                                                 out + (size_t)TSTEPS * NH, (long long)RH);
    }
}
